// GridDAFusion_87290915324622
// MI455X (gfx1250) — matrix operand order repaired and checked by reading, not compiled, not run on hardware
//
#include <hip/hip_runtime.h>
#include <math.h>

// Problem constants (from reference): B=8, C=64, H=W=256, N=128, K=(2*16+1)^2=1089
#define Hh 256
#define Ww 256
#define HW 65536
#define Bn 8
#define Cc 64
#define Nn 128
#define KK 1089
#define KPAD 1104      // 69 tiles * 16 edges
#define NTILES 69
#define HID 32

typedef __attribute__((ext_vector_type(16))) _Float16 v16h;
typedef __attribute__((ext_vector_type(8)))  float    v8f;

union V16H { v16h v; _Float16 h[16]; unsigned u[8]; };

__device__ __forceinline__ float relu_(float x) { return fmaxf(x, 0.f); }

__device__ __forceinline__ unsigned pack2h(float a, float b) {
  union { _Float16 h[2]; unsigned u; } x;
  x.h[0] = (_Float16)a; x.h[1] = (_Float16)b;
  return x.u;
}

// ordered-int float atomic max (works for mixed-sign values, init = -1e9)
__device__ __forceinline__ void atomicMaxF(float* addr, float v) {
  if (v >= 0.f) atomicMax((int*)addr, __float_as_int(v));
  else          atomicMin((unsigned int*)addr, __float_as_uint(v));
}

// ---------------------------------------------------------------------------
// K0: per-source MLPs.  h_pay = MLP2(pay); ctxc = att_b1 + MLP2(ctx) @ att_w1[32:64]
// ---------------------------------------------------------------------------
__global__ void k0_sources(const float* __restrict__ ctx, const float* __restrict__ pay,
    const float* __restrict__ cw1, const float* __restrict__ cb1,
    const float* __restrict__ cw2, const float* __restrict__ cb2,
    const float* __restrict__ pw1, const float* __restrict__ pb1,
    const float* __restrict__ pw2, const float* __restrict__ pb2,
    const float* __restrict__ aw1, const float* __restrict__ ab1,
    float* __restrict__ hpay, float* __restrict__ ctxc)
{
  int sid = blockIdx.x * blockDim.x + threadIdx.x;
  if (sid >= Bn * Nn) return;
  float x[32], h1[32], h2[32];
  for (int i = 0; i < 32; ++i) x[i] = ctx[sid * 32 + i];
  for (int o = 0; o < 32; ++o) {
    float s = cb1[o];
    for (int i = 0; i < 32; ++i) s += x[i] * cw1[i * 32 + o];
    h1[o] = relu_(s);
  }
  for (int o = 0; o < 32; ++o) {
    float s = cb2[o];
    for (int i = 0; i < 32; ++i) s += h1[i] * cw2[i * 32 + o];
    h2[o] = relu_(s);
  }
  for (int o = 0; o < 32; ++o) {
    float s = ab1[o];
    for (int i = 0; i < 32; ++i) s += h2[i] * aw1[(32 + i) * 32 + o];
    ctxc[sid * 32 + o] = s;
  }
  float y[16];
  for (int i = 0; i < 16; ++i) y[i] = pay[sid * 16 + i];
  for (int o = 0; o < 32; ++o) {
    float s = pb1[o];
    for (int i = 0; i < 16; ++i) s += y[i] * pw1[i * 32 + o];
    h1[o] = relu_(s);
  }
  for (int o = 0; o < 32; ++o) {
    float s = pb2[o];
    for (int i = 0; i < 32; ++i) s += h1[i] * pw2[i * 32 + o];
    hpay[sid * 32 + o] = relu_(s);
  }
}

// ---------------------------------------------------------------------------
// Init kernels (harness poisons buffers; we must fully initialize each call)
// ---------------------------------------------------------------------------
__global__ void p0_init_out(float* __restrict__ out) {
  size_t i = (size_t)blockIdx.x * blockDim.x + threadIdx.x;
  if (i < (size_t)Bn * HID * HW) out[i] = 0.f;
}
__global__ void p0_init_ws(float* __restrict__ pixmax, float* __restrict__ den) {
  int i = blockIdx.x * blockDim.x + threadIdx.x;
  if (i < Bn * HW) { pixmax[i] = -1e9f; den[i] = 0.f; }
}

// ---------------------------------------------------------------------------
// P1: per-edge scores via WMMA.  One wave32 per source; 16 edges per tile.
//   hq[32 x 16]  = reluWMMA( q_w^T[32x64] x Fgather[64x16] + q_b )
//   z[32 x 16]   = WMMA( att_w1_hq^T[32x32] x hq_f16[32x16] + (ctxc+geom) )
//   score[16]    = relu(z)·att_w2 + att_b2   (lane-pair reduction)
// ---------------------------------------------------------------------------
__global__ __launch_bounds__(256) void p1_scores(
    const float* __restrict__ F, const int* __restrict__ rows, const int* __restrict__ cols,
    const unsigned char* __restrict__ valid, const float* __restrict__ elev,
    const float* __restrict__ selev, const float* __restrict__ q_w,
    const float* __restrict__ q_b, const float* __restrict__ att_w1,
    const float* __restrict__ att_w2, const float* __restrict__ att_b2,
    const float* __restrict__ ctxc, float* __restrict__ scores, float* __restrict__ pixmax)
{
  const int lane = threadIdx.x & 31;
  const int sid  = blockIdx.x * (blockDim.x >> 5) + (threadIdx.x >> 5);
  if (sid >= Bn * Nn) return;
  const int half = lane >> 4;     // K-half selector in WMMA layouts
  const int colL = lane & 15;     // M (A rows) / N (B,C,D columns)
  const int b = sid >> 7;

  // --- A-matrices (weights transposed), 16-bit A operand, this code's contiguous k order:
  //     lane(0-15)=M, element t -> K = t + 16*half
  V16H aq00, aq01, aq10, aq11, aw0, aw1t;
#pragma unroll
  for (int t = 0; t < 16; ++t) {
    int K = t + 16 * half;
    aq00.h[t] = (_Float16)q_w[K * HID + colL];           // M chunk 0, feat 0..31
    aq10.h[t] = (_Float16)q_w[K * HID + 16 + colL];      // M chunk 1
    aq01.h[t] = (_Float16)q_w[(32 + K) * HID + colL];    // feat 32..63
    aq11.h[t] = (_Float16)q_w[(32 + K) * HID + 16 + colL];
    aw0.h[t]  = (_Float16)att_w1[K * HID + colL];        // att_w1 rows 0..31 (hq)
    aw1t.h[t] = (_Float16)att_w1[K * HID + 16 + colL];
  }

  // --- per-lane constants for the 8 C/D rows this lane owns in each M-tile
  float qb0[8], qb1[8], ca0[8], ca1[8], w20[8], w21[8], g0[4][8], g1[4][8];
#pragma unroll
  for (int j = 0; j < 8; ++j) {
    int og0 = 8 * half + j, og1 = 16 + og0;
    qb0[j] = q_b[og0];                     qb1[j] = q_b[og1];
    ca0[j] = ctxc[sid * HID + og0];        ca1[j] = ctxc[sid * HID + og1];
    w20[j] = att_w2[og0];                  w21[j] = att_w2[og1];
#pragma unroll
    for (int g = 0; g < 4; ++g) {
      g0[g][j] = att_w1[(64 + g) * HID + og0];
      g1[g][j] = att_w1[(64 + g) * HID + og1];
    }
  }
  const int   srow = rows[sid], scol = cols[sid];
  const bool  vld  = valid[sid] != 0;
  const float se   = selev[sid];
  const float ab2  = att_b2[0];
  const float* Fb  = F    + (size_t)b * Cc * HW;
  const float* eb  = elev + (size_t)b * HW;
  float*       pm  = pixmax + (size_t)b * HW;

  for (int tile = 0; tile < NTILES; ++tile) {
    const int k  = tile * 16 + colL;
    const int dr = k / 33 - 16, dc = k % 33 - 16;
    const int qr = srow + dr,   qc = scol + dc;
    const int d2 = dr * dr + dc * dc;
    const bool mask = vld && (k < KK) && (d2 <= 256) &&
                      (qr >= 0) && (qr < Hh) && (qc >= 0) && (qc < Ww);
    const int pix = min(max(qr, 0), Hh - 1) * Ww + min(max(qc, 0), Ww - 1);

    // --- gather features into B layout: lane column = colL, K = t + 16*half
    V16H bf0, bf1;
#pragma unroll
    for (int t = 0; t < 16; ++t) {
      int K = t + 16 * half;
      bf0.h[t] = (_Float16)Fb[(size_t)K * HW + pix];
      bf1.h[t] = (_Float16)Fb[(size_t)(32 + K) * HW + pix];
    }

    // --- hq = q_w^T x F + q_b   (4 WMMAs: 2 K-chunks x 2 M-chunks)
    v8f acc0, acc1;
#pragma unroll
    for (int j = 0; j < 8; ++j) { acc0[j] = qb0[j]; acc1[j] = qb1[j]; }
    acc0 = __builtin_amdgcn_wmma_f32_16x16x32_f16(false, aq00.v, false, bf0.v, (short)0, acc0, false, false);
    acc0 = __builtin_amdgcn_wmma_f32_16x16x32_f16(false, aq01.v, false, bf1.v, (short)0, acc0, false, false);
    acc1 = __builtin_amdgcn_wmma_f32_16x16x32_f16(false, aq10.v, false, bf0.v, (short)0, acc1, false, false);
    acc1 = __builtin_amdgcn_wmma_f32_16x16x32_f16(false, aq11.v, false, bf1.v, (short)0, acc1, false, false);

    // --- relu + f32->f16 pack; D-layout -> B-layout needs one shfl_xor(16)
    unsigned p0[4], p1[4], s0[4], s1[4];
#pragma unroll
    for (int w = 0; w < 4; ++w) {
      p0[w] = pack2h(relu_(acc0[2 * w]), relu_(acc0[2 * w + 1]));
      p1[w] = pack2h(relu_(acc1[2 * w]), relu_(acc1[2 * w + 1]));
    }
#pragma unroll
    for (int w = 0; w < 4; ++w) {
      s0[w] = (unsigned)__shfl_xor((int)p0[w], 16, 32);
      s1[w] = (unsigned)__shfl_xor((int)p1[w], 16, 32);
    }
    V16H bh;
#pragma unroll
    for (int w = 0; w < 4; ++w) {
      bh.u[w]     = half ? s1[w] : p0[w];   // K 0..7  | 16..23
      bh.u[4 + w] = half ? p1[w] : s0[w];   // K 8..15 | 24..31
    }

    // --- geometry (+ constant ctx part) seeds the attention accumulator.
    // bearing = atan2(dc,-dr): sin(b) = dc/dist, cos(b) = -dr/dist (exact),
    // with the dist==0 limit sin=0, cos=-1 (atan2(0,-0)=pi).  Avoids the
    // expensive atan2f/sinf/cosf software expansions in the WMMA hot loop.
    const float fdr = (float)dr, fdc = (float)dc;
    const float dist = sqrtf((float)d2);
    const float inv = (d2 > 0) ? (1.f / dist) : 0.f;
    const float sb = fdc * inv;
    const float cb = (d2 > 0) ? (-fdr * inv) : -1.f;
    const float dn = fmaxf(dist, 1e-6f) * (1.f / 16.f);
    const float delta = (eb[pix] - se) * 1e-3f;

    v8f z0, z1;
#pragma unroll
    for (int j = 0; j < 8; ++j) {
      z0[j] = ca0[j] + dn * g0[0][j] + sb * g0[1][j] + cb * g0[2][j] + delta * g0[3][j];
      z1[j] = ca1[j] + dn * g1[0][j] + sb * g1[1][j] + cb * g1[2][j] + delta * g1[3][j];
    }
    z0 = __builtin_amdgcn_wmma_f32_16x16x32_f16(false, aw0.v,  false, bh.v, (short)0, z0, false, false);
    z1 = __builtin_amdgcn_wmma_f32_16x16x32_f16(false, aw1t.v, false, bh.v, (short)0, z1, false, false);

    // --- score = relu(z)·att_w2 + b2; each lane holds 16 of 32 dims, partner
    //     lane (lane^16) holds the rest
    float part = 0.f;
#pragma unroll
    for (int j = 0; j < 8; ++j) part += relu_(z0[j]) * w20[j] + relu_(z1[j]) * w21[j];
    const float sc = part + __shfl_xor(part, 16, 32) + ab2;
    const float sv = mask ? sc : -1e9f;
    if (half == 0) {
      scores[(size_t)sid * KPAD + k] = sv;
      if (mask) atomicMaxF(pm + pix, sc);
    }
  }
}

// ---------------------------------------------------------------------------
// P2: num = exp(s - max[pix]) * mask; den[pix] += num  (num overwrites score)
// ---------------------------------------------------------------------------
__global__ void p2_softmax_den(const int* __restrict__ rows, const int* __restrict__ cols,
    const unsigned char* __restrict__ valid, float* __restrict__ scores,
    const float* __restrict__ pixmax, float* __restrict__ den)
{
  long long e = (long long)blockIdx.x * blockDim.x + threadIdx.x;
  if (e >= (long long)Bn * Nn * KK) return;
  const int k = (int)(e % KK);
  const int n = (int)((e / KK) % Nn);
  const int b = (int)(e / ((long long)KK * Nn));
  const int sid = b * Nn + n;
  const int dr = k / 33 - 16, dc = k % 33 - 16;
  const int qr = rows[sid] + dr, qc = cols[sid] + dc;
  const bool mask = (valid[sid] != 0) && (dr * dr + dc * dc <= 256) &&
                    (qr >= 0) && (qr < Hh) && (qc >= 0) && (qc < Ww);
  const int pix = min(max(qr, 0), Hh - 1) * Ww + min(max(qc, 0), Ww - 1);
  const float s = scores[(size_t)sid * KPAD + k];
  float num = 0.f;
  if (mask) num = expf(s - pixmax[b * HW + pix]);
  scores[(size_t)sid * KPAD + k] = num;
  if (num > 0.f) atomicAdd(&den[b * HW + pix], num);
}

// ---------------------------------------------------------------------------
// P3: alpha = num/den; scatter-add alpha * h_pay[32] into da[b,:,pix]
// ---------------------------------------------------------------------------
__global__ void p3_scatter(const int* __restrict__ rows, const int* __restrict__ cols,
    const float* __restrict__ scores, const float* __restrict__ den,
    const float* __restrict__ hpay, float* __restrict__ out)
{
  long long e = (long long)blockIdx.x * blockDim.x + threadIdx.x;
  if (e >= (long long)Bn * Nn * KK) return;
  const int k = (int)(e % KK);
  const int n = (int)((e / KK) % Nn);
  const int b = (int)(e / ((long long)KK * Nn));
  const int sid = b * Nn + n;
  const float num = scores[(size_t)sid * KPAD + k];
  if (num <= 0.f) return;  // num > 0 implies a real (masked-in) edge
  const int dr = k / 33 - 16, dc = k % 33 - 16;
  const int qr = rows[sid] + dr, qc = cols[sid] + dc;
  const int pix = min(max(qr, 0), Hh - 1) * Ww + min(max(qc, 0), Ww - 1);
  const float alpha = num / fmaxf(den[b * HW + pix], 1e-20f);
  const float* hp = hpay + (size_t)sid * HID;
  float* dst = out + (size_t)b * HID * HW + pix;
#pragma unroll
  for (int h = 0; h < HID; ++h) atomicAdd(dst + (size_t)h * HW, alpha * hp[h]);
}

// ---------------------------------------------------------------------------
// P4: coverage = (den > 0)
// ---------------------------------------------------------------------------
__global__ void p4_cov(const float* __restrict__ den, float* __restrict__ out) {
  int i = blockIdx.x * blockDim.x + threadIdx.x;
  if (i < Bn * HW) out[(size_t)Bn * HID * HW + i] = (den[i] > 0.f) ? 1.f : 0.f;
}

// ---------------------------------------------------------------------------
extern "C" void kernel_launch(void* const* d_in, const int* in_sizes, int n_in,
                              void* d_out, int out_size, void* d_ws, size_t ws_size,
                              hipStream_t stream) {
  (void)in_sizes; (void)n_in; (void)out_size; (void)ws_size;
  const float*         F       = (const float*)d_in[0];
  const int*           rows    = (const int*)d_in[1];
  const int*           cols    = (const int*)d_in[2];
  const float*         ctx     = (const float*)d_in[3];
  const float*         pay     = (const float*)d_in[4];
  const unsigned char* valid   = (const unsigned char*)d_in[5];
  const float*         elev    = (const float*)d_in[6];
  const float*         selev   = (const float*)d_in[7];
  const float *cw1 = (const float*)d_in[8],  *cb1 = (const float*)d_in[9];
  const float *cw2 = (const float*)d_in[10], *cb2 = (const float*)d_in[11];
  const float *pw1 = (const float*)d_in[12], *pb1 = (const float*)d_in[13];
  const float *pw2 = (const float*)d_in[14], *pb2 = (const float*)d_in[15];
  const float *qw  = (const float*)d_in[16], *qb  = (const float*)d_in[17];
  const float *aw1 = (const float*)d_in[18], *ab1 = (const float*)d_in[19];
  const float *aw2 = (const float*)d_in[20], *ab2 = (const float*)d_in[21];
  float* out = (float*)d_out;

  // workspace layout (floats)
  float* ws      = (float*)d_ws;
  float* scores  = ws;                                   // B*N*KPAD = 1,130,496
  float* pixmax  = scores + (size_t)Bn * Nn * KPAD;      // B*HW     =   524,288
  float* den     = pixmax + (size_t)Bn * HW;             // B*HW
  float* hpay    = den    + (size_t)Bn * HW;             // B*N*32
  float* ctxc    = hpay   + (size_t)Bn * Nn * HID;       // B*N*32

  const long long nEdges = (long long)Bn * Nn * KK;      // 1,115,136

  p0_init_out<<<(int)(((size_t)Bn * HID * HW + 255) / 256), 256, 0, stream>>>(out);
  p0_init_ws<<<(Bn * HW + 255) / 256, 256, 0, stream>>>(pixmax, den);
  k0_sources<<<(Bn * Nn + 255) / 256, 256, 0, stream>>>(ctx, pay, cw1, cb1, cw2, cb2,
                                                        pw1, pb1, pw2, pb2, aw1, ab1,
                                                        hpay, ctxc);
  // 8 waves/block, one source per wave -> 1024 waves total
  p1_scores<<<(Bn * Nn) / 8, 256, 0, stream>>>(F, rows, cols, valid, elev, selev,
                                               qw, qb, aw1, aw2, ab2, ctxc,
                                               scores, pixmax);
  p2_softmax_den<<<(int)((nEdges + 255) / 256), 256, 0, stream>>>(rows, cols, valid,
                                                                  scores, pixmax, den);
  p3_scatter<<<(int)((nEdges + 255) / 256), 256, 0, stream>>>(rows, cols, scores, den,
                                                              hpay, out);
  p4_cov<<<(Bn * HW + 255) / 256, 256, 0, stream>>>(den, out);
}
